// RelationTypeHead_20942260535549
// MI455X (gfx1250) — matrix operand order repaired and checked by reading, not compiled, not run on hardware
//
#include <hip/hip_runtime.h>
#include <hip/hip_bf16.h>

// ---------------------------------------------------------------------------
// RelationTypeHead on MI455X (gfx1250): all contractions via
// v_wmma_f32_16x16x32_f16 (f16 in, f32 accumulate).
//
// Shapes: B=16, N=256, H=768, P=2048, R=3.  BN = 4096.
// Stage 0: convert feat/head_W/tail_W f32->f16; transpose-convert bil_W to
//          bilWT[r][q][p] so every GEMM reads B^T tiles row-major.
// Stage 1: head16 = feat16 @ head_W^T + head_b  (M=4096,N=2048,K=768), same tail.
// Stage 2: hw16[r] = head16 @ W[r]              (M=4096,N=2048,K=2048), r=0..2.
// Stage 3: logits[b,n,m,r] = hw16[r][b,n,:] . tail16[b,m,:] + bil_b[r]
//          (48 GEMMs of 256x256x2048), f32 out.
// ---------------------------------------------------------------------------

typedef _Float16 half_t;
typedef __attribute__((ext_vector_type(16))) _Float16 v16h;
typedef __attribute__((ext_vector_type(8)))  _Float16 v8h;
typedef __attribute__((ext_vector_type(4)))  _Float16 v4h;
typedef __attribute__((ext_vector_type(8)))  float    v8f;
typedef __attribute__((ext_vector_type(4)))  float    v4f;

typedef union { v16h v; v8h h[2]; } frag16;

#define TILE_M 128
#define TILE_N 128
#define TILE_K 32
#define LDSW   40   // halves per LDS row: 32 + 8 pad (80B) to spread banks

// ---------------------------------------------------------------------------
// f32 -> f16 elementwise convert, 4 elements/thread (n must be multiple of 4)
// ---------------------------------------------------------------------------
__global__ __launch_bounds__(256) void k_cvt_f16(const float* __restrict__ in,
                                                 half_t* __restrict__ out, int n4) {
  int i = blockIdx.x * 256 + threadIdx.x;
  if (i < n4) {
    v4f x = ((const v4f*)in)[i];
    ((v4h*)out)[i] = __builtin_convertvector(x, v4h);
  }
}

// ---------------------------------------------------------------------------
// bil_W[r][p][q] f32  ->  bilWT[r][q][p] f16   (2048x2048 per r, 64x64 tiles)
// ---------------------------------------------------------------------------
__global__ __launch_bounds__(256) void k_tr_cvt(const float* __restrict__ in,
                                                half_t* __restrict__ out) {
  __shared__ half_t tile[64][68];
  const int r  = blockIdx.z;
  const int p0 = blockIdx.x * 64;
  const int q0 = blockIdx.y * 64;
  const float*  src = in  + (size_t)r * 2048 * 2048;
  half_t*       dst = out + (size_t)r * 2048 * 2048;
  const int tx  = threadIdx.x & 63;
  const int ty0 = (threadIdx.x >> 6) * 16;
#pragma unroll
  for (int j = 0; j < 16; ++j) {
    int ty = ty0 + j;
    tile[ty][tx] = (half_t)src[(size_t)(p0 + ty) * 2048 + q0 + tx];
  }
  __syncthreads();
#pragma unroll
  for (int j = 0; j < 16; ++j) {
    int ty = ty0 + j;
    dst[(size_t)(q0 + ty) * 2048 + p0 + tx] = tile[tx][ty];
  }
}

// ---------------------------------------------------------------------------
// Shared GEMM mainloop: C_tile(128x128) += A(128xK) * Bt(128xK)^T
//   A : row-major [M][K] f16, pre-offset to tile row base
//   Bt: row-major [N][K] f16 (i.e. B^T), pre-offset to tile col base
// 256 threads = 8 waves; wave (wr,wc) in 2x4 grid owns 64x32 patch
// (4x2 WMMA accumulators). Double-buffered LDS, register prefetch.
// ---------------------------------------------------------------------------
__device__ __forceinline__ void gemm_mainloop(const half_t* __restrict__ A, int lda,
                                              const half_t* __restrict__ Bt, int ldb,
                                              int K, v8f acc[4][2]) {
  __shared__ half_t sA[2][TILE_M * LDSW];
  __shared__ half_t sB[2][TILE_N * LDSW];

  const int tid  = threadIdx.x;
  const int lane = tid & 31;
  const int wave = tid >> 5;
  const int wr   = wave >> 2;          // 0..1 (row of wave grid)
  const int wc   = wave & 3;           // 0..3 (col of wave grid)
  const int m16  = lane & 15;
  const int lh   = lane >> 4;          // lane half (0/1)

  // global->LDS: thread t loads 16 contiguous halves (32B) of one row
  const int lrow = tid >> 1;           // 0..127
  const int lseg = (tid & 1) * 16;     // 0 or 16 halves within the 32-half row
  const half_t* Ag = A + (size_t)lrow * lda + lseg;
  const half_t* Bg = Bt + (size_t)lrow * ldb + lseg;
  const int lidx = lrow * LDSW + lseg;

  v8h ra0 = *(const v8h*)(Ag);
  v8h ra1 = *(const v8h*)(Ag + 8);
  v8h rb0 = *(const v8h*)(Bg);
  v8h rb1 = *(const v8h*)(Bg + 8);

  const int nk = K >> 5;
  for (int kt = 0; kt < nk; ++kt) {
    half_t* a = sA[kt & 1];
    half_t* b = sB[kt & 1];
    *(v8h*)(a + lidx)     = ra0;
    *(v8h*)(a + lidx + 8) = ra1;
    *(v8h*)(b + lidx)     = rb0;
    *(v8h*)(b + lidx + 8) = rb1;
    __syncthreads();

    if (kt + 1 < nk) {  // prefetch next K-slab while WMMAs run
      const half_t* an = Ag + (kt + 1) * TILE_K;
      const half_t* bn = Bg + (kt + 1) * TILE_K;
      ra0 = *(const v8h*)(an);
      ra1 = *(const v8h*)(an + 8);
      rb0 = *(const v8h*)(bn);
      rb1 = *(const v8h*)(bn + 8);
    }

    frag16 af[4], bf[2];
    // A 16x32 f16 layout: lane m = lane%16; VGPR0-3 hold K = lh*8..lh*8+7,
    // VGPR4-7 hold K = 16+lh*8.. -> two 16B LDS loads per fragment.
#pragma unroll
    for (int mi = 0; mi < 4; ++mi) {
      const half_t* p = a + (wr * 64 + mi * 16 + m16) * LDSW + lh * 8;
      af[mi].h[0] = *(const v8h*)(p);
      af[mi].h[1] = *(const v8h*)(p + 16);
    }
    // B 32x16 f16 layout: lane n = lane%16; VGPR0-3 hold K = lh*8..lh*8+7,
    // VGPR4-7 hold K = 16+lh*8.. -> two 16B LDS loads per fragment.
#pragma unroll
    for (int ni = 0; ni < 2; ++ni) {
      const half_t* p = b + (wc * 32 + ni * 16 + m16) * LDSW + lh * 8;
      bf[ni].h[0] = *(const v8h*)(p);
      bf[ni].h[1] = *(const v8h*)(p + 16);
    }
#pragma unroll
    for (int mi = 0; mi < 4; ++mi)
#pragma unroll
      for (int ni = 0; ni < 2; ++ni)
        acc[mi][ni] = __builtin_amdgcn_wmma_f32_16x16x32_f16(
            false, af[mi].v, false, bf[ni].v, (short)0, acc[mi][ni], false, false);
    __syncthreads();
  }
}

// ---------------------------------------------------------------------------
// Stages 1 & 2: f16 output, optional per-column bias, optional z batching
// ---------------------------------------------------------------------------
__global__ __launch_bounds__(256) void k_gemm_h16(
    const half_t* __restrict__ A, int lda, long sAz,
    const half_t* __restrict__ Bt, int ldb, long sBz,
    const float* __restrict__ bias,
    half_t* __restrict__ C, int ldc, long sCz, int K) {
  const int rowBase = blockIdx.x * TILE_M;
  const int colBase = blockIdx.y * TILE_N;
  A  += (size_t)blockIdx.z * sAz + (size_t)rowBase * lda;
  Bt += (size_t)blockIdx.z * sBz + (size_t)colBase * ldb;
  C  += (size_t)blockIdx.z * sCz;

  v8f acc[4][2] = {};
  gemm_mainloop(A, lda, Bt, ldb, K, acc);

  const int lane = threadIdx.x & 31, wave = threadIdx.x >> 5;
  const int wr = wave >> 2, wc = wave & 3;
  const int m16 = lane & 15, lh = lane >> 4;
#pragma unroll
  for (int mi = 0; mi < 4; ++mi)
#pragma unroll
    for (int ni = 0; ni < 2; ++ni) {
      const int col = colBase + wc * 32 + ni * 16 + m16;
      const float bv = bias ? bias[col] : 0.0f;
#pragma unroll
      for (int i = 0; i < 8; ++i) {
        const int row = rowBase + wr * 64 + mi * 16 + lh * 8 + i;
        C[(size_t)row * ldc + col] = (half_t)(acc[mi][ni][i] + bv);
      }
    }
}

// ---------------------------------------------------------------------------
// Stage 3: logits[b,n,m,r] = hw16[r][b*256+n,:] . tail16[b*256+m,:] + bil_b[r]
// grid.z = b*3 + r (48), grid.x/y = 2x2 tiles of the 256x256 output
// ---------------------------------------------------------------------------
__global__ __launch_bounds__(256) void k_gemm_logits(
    const half_t* __restrict__ hw, const half_t* __restrict__ tail,
    const float* __restrict__ bilb, float* __restrict__ out) {
  const int z = blockIdx.z;
  const int b = z / 3, r = z % 3;
  const int rowBase = blockIdx.x * TILE_M;
  const int colBase = blockIdx.y * TILE_N;
  const half_t* A  = hw + (size_t)r * 4096 * 2048 + (size_t)b * 256 * 2048
                        + (size_t)rowBase * 2048;
  const half_t* Bt = tail + (size_t)b * 256 * 2048 + (size_t)colBase * 2048;

  v8f acc[4][2] = {};
  gemm_mainloop(A, 2048, Bt, 2048, 2048, acc);

  const float bb = bilb[r];
  float* o = out + (size_t)b * 256 * 256 * 3 + r;   // [b][n][m][r]
  const int lane = threadIdx.x & 31, wave = threadIdx.x >> 5;
  const int wr = wave >> 2, wc = wave & 3;
  const int m16 = lane & 15, lh = lane >> 4;
#pragma unroll
  for (int mi = 0; mi < 4; ++mi)
#pragma unroll
    for (int ni = 0; ni < 2; ++ni) {
      const int col = colBase + wc * 32 + ni * 16 + m16;       // m
#pragma unroll
      for (int i = 0; i < 8; ++i) {
        const int row = rowBase + wr * 64 + mi * 16 + lh * 8 + i;  // n
        o[((size_t)row * 256 + col) * 3] = acc[mi][ni][i] + bb;
      }
    }
}

// ---------------------------------------------------------------------------
extern "C" void kernel_launch(void* const* d_in, const int* in_sizes, int n_in,
                              void* d_out, int out_size, void* d_ws, size_t ws_size,
                              hipStream_t stream) {
  const float* feat  = (const float*)d_in[0];   // [16,256,768]
  const float* headW = (const float*)d_in[1];   // [2048,768]
  const float* headb = (const float*)d_in[2];   // [2048]
  const float* tailW = (const float*)d_in[3];   // [2048,768]
  const float* tailb = (const float*)d_in[4];   // [2048]
  const float* bilW  = (const float*)d_in[5];   // [3,2048,2048]
  const float* bilb  = (const float*)d_in[6];   // [3]
  float* out = (float*)d_out;                   // [16,256,256,3]

  // f16 workspace carve-up (element counts)
  half_t* ws       = (half_t*)d_ws;
  half_t* feat16   = ws;                        // 4096*768   = 3,145,728
  half_t* headW16  = feat16  + 3145728;         // 2048*768   = 1,572,864
  half_t* tailW16  = headW16 + 1572864;         // 2048*768
  half_t* bilWT16  = tailW16 + 1572864;         // 3*2048*2048 = 12,582,912
  half_t* head16   = bilWT16 + 12582912;        // 4096*2048  = 8,388,608
  half_t* tail16   = head16  + 8388608;         // 4096*2048
  half_t* hw16     = tail16  + 8388608;         // 3*4096*2048 = 25,165,824

  // Stage 0: conversions
  k_cvt_f16<<<(3145728 / 4 + 255) / 256, 256, 0, stream>>>(feat,  feat16,  3145728 / 4);
  k_cvt_f16<<<(1572864 / 4 + 255) / 256, 256, 0, stream>>>(headW, headW16, 1572864 / 4);
  k_cvt_f16<<<(1572864 / 4 + 255) / 256, 256, 0, stream>>>(tailW, tailW16, 1572864 / 4);
  k_tr_cvt<<<dim3(32, 32, 3), 256, 0, stream>>>(bilW, bilWT16);

  // Stage 1: head/tail projections (M=4096, N=2048, K=768)
  dim3 g1(4096 / TILE_M, 2048 / TILE_N, 1);
  k_gemm_h16<<<g1, 256, 0, stream>>>(feat16, 768, 0, headW16, 768, 0,
                                     headb, head16, 2048, 0, 768);
  k_gemm_h16<<<g1, 256, 0, stream>>>(feat16, 768, 0, tailW16, 768, 0,
                                     tailb, tail16, 2048, 0, 768);

  // Stage 2: hw[r] = head @ W[r] (M=4096, N=2048, K=2048), z = r
  dim3 g2(4096 / TILE_M, 2048 / TILE_N, 3);
  k_gemm_h16<<<g2, 256, 0, stream>>>(head16, 2048, 0,
                                     bilWT16, 2048, 2048L * 2048,
                                     nullptr, hw16, 2048, 4096L * 2048, 2048);

  // Stage 3: logits (48 x [256x256x2048]), z = b*3 + r
  dim3 g3(256 / TILE_M, 256 / TILE_N, 48);
  k_gemm_logits<<<g3, 256, 0, stream>>>(hw16, tail16, bilb, out);
}
